// MSTGCN_block_39840116638587
// MI455X (gfx1250) — matrix operand order repaired and checked by reading, not compiled, not run on hardware
//
#include <hip/hip_runtime.h>
#include <hip/hip_bf16.h>

// ---------------- problem constants ----------------
constexpr int kB = 2, kN = 10000, kF = 32, kT = 16, kE = 160000, kC = 64;
constexpr int kNodeF = kF * kT;   // 512 floats per (b, node) input block
constexpr int kNodeC = kC * kT;   // 1024 values per (b, node) channel block
constexpr float kEps = 1e-5f;

typedef _Float16 half16 __attribute__((ext_vector_type(16)));
typedef float    float8 __attribute__((ext_vector_type(8)));

__device__ __forceinline__ float8 wmma_f16(half16 a, half16 b, float8 c) {
  // D = A(16x32 f16) * B(32x16 f16) + C(16x16 f32)
  return __builtin_amdgcn_wmma_f32_16x16x32_f16(false, a, false, b, (short)0, c,
                                                false, false);
}

// 16-bit A-matrix 16x32 per-lane k mapping (contiguous, matches B fills):
// lane<16: k = {0..15}; lane>=16: k = {16..31}
__device__ __forceinline__ int kmapA(int lane, int j) {
  return ((lane >= 16) ? 16 : 0) + j;
}

// ---------------- small prep kernels ----------------
__global__ void zero_kernel(float* __restrict__ p, long n) {
  long i = (long)blockIdx.x * blockDim.x + threadIdx.x;
  long s = (long)gridDim.x * blockDim.x;
  for (; i < n; i += s) p[i] = 0.0f;
}

__global__ void deg_kernel(const int* __restrict__ row, float* __restrict__ deg) {
  int e = blockIdx.x * blockDim.x + threadIdx.x;
  if (e < kE) atomicAdd(&deg[row[e]], 1.0f);
}

__global__ void dinv_kernel(const float* __restrict__ deg, float* __restrict__ dinv) {
  int i = blockIdx.x * blockDim.x + threadIdx.x;
  if (i < kN) {
    float d = deg[i];
    dinv[i] = (d > 0.0f) ? rsqrtf(d) : 0.0f;
  }
}

__global__ void edgew_kernel(const int* __restrict__ ei, const float* __restrict__ dinv,
                             float* __restrict__ w) {
  int e = blockIdx.x * blockDim.x + threadIdx.x;
  if (e < kE) w[e] = -(dinv[ei[e]] * dinv[ei[kE + e]]);
}

// ---- pack weights into per-lane WMMA A-fragment order (coalesced 32B/lane loads) ----
__global__ void pack_cheb(const float* __restrict__ W, _Float16* __restrict__ frag) {
  int idx = blockIdx.x * blockDim.x + threadIdx.x;      // [k][mtile][lane][j]
  if (idx >= 3 * 4 * 32 * 16) return;
  int j = idx & 15, lane = (idx >> 4) & 31, m = (idx >> 9) & 3, k = idx >> 11;
  int c = m * 16 + (lane & 15);
  int f = kmapA(lane, j);
  frag[idx] = (_Float16)W[(k * kF + f) * kC + c];       // cheb_W (K,F,C)
}

__global__ void pack_time(const float* __restrict__ W, _Float16* __restrict__ frag) {
  int idx = blockIdx.x * blockDim.x + threadIdx.x;      // [dt][ktile][mtile][lane][j]
  if (idx >= 3 * 2 * 4 * 32 * 16) return;
  int j = idx & 15, lane = (idx >> 4) & 31, m = (idx >> 9) & 3;
  int kt = (idx >> 11) & 1, dt = idx >> 12;
  int co = m * 16 + (lane & 15);
  int ci = kt * 32 + kmapA(lane, j);
  frag[idx] = (_Float16)W[(co * kC + ci) * 3 + dt];     // time_W (Co,Ci,1,3)
}

__global__ void pack_res(const float* __restrict__ W, _Float16* __restrict__ frag) {
  int idx = blockIdx.x * blockDim.x + threadIdx.x;      // [mtile][lane][j]
  if (idx >= 4 * 32 * 16) return;
  int j = idx & 15, lane = (idx >> 4) & 31, m = idx >> 9;
  int co = m * 16 + (lane & 15);
  int f = kmapA(lane, j);
  frag[idx] = (_Float16)W[co * kF + f];                 // res_W (Co,F,1,1)
}

// ---------------- graph propagation: agg[b,col] += w * src[b,row] ----------------
// src/dst tensors are 41 MB each -> both live in the 192 MB L2; the f32
// atomic-scatter resolves at L2 atomic units, gathers are 64B-coalesced.
__global__ void prop_kernel(const float* __restrict__ src, float* __restrict__ dst,
                            const float* __restrict__ w, const int* __restrict__ row,
                            const int* __restrict__ col) {
  int e = blockIdx.x;
  float we = w[e];
  if (we == 0.0f) return;                 // isolated endpoints contribute nothing
  int r = row[e], c = col[e];
  int tid = threadIdx.x;                  // 256 threads = 2 batches * 128 float4
  int b = tid >> 7;
  int q = tid & 127;
  const float4* s = (const float4*)(src + ((size_t)(b * kN + r)) * kNodeF) + q;
  float* d = dst + ((size_t)(b * kN + c)) * kNodeF + (size_t)q * 4;
  float4 v = *s;
  atomicAdd(d + 0, we * v.x);
  atomicAdd(d + 1, we * v.y);
  atomicAdd(d + 2, we * v.z);
  atomicAdd(d + 3, we * v.w);
}

// ---------------- Cheb einsum + bias + ReLU, fused, WMMA ----------------
// per wave: one (b,n); out(64x16) = sum_k Wk^T(64x32) @ Txk(32x16); f32 accum.
__global__ void __launch_bounds__(128) cheb_wmma_kernel(
    const float* __restrict__ x, const float* __restrict__ tx1,
    const float* __restrict__ tx2raw, const _Float16* __restrict__ afrag,
    const float* __restrict__ chebB, _Float16* __restrict__ spatial) {
  int wid = threadIdx.x >> 5, lane = threadIdx.x & 31;
  int node = blockIdx.x * 4 + wid;        // B*N == 20000, exact
  int b = node / kN, n = node % kN;
  size_t base = ((size_t)b * kN + n) * kNodeF;

  int t = lane & 15;
  int fhi = (lane >= 16) ? 16 : 0;        // B-matrix 32x16: lane half selects k-half
  half16 B0, B1, B2;
#pragma unroll
  for (int j = 0; j < 16; ++j) {          // per j: two contiguous 64B rows -> coalesced
    size_t a = base + (size_t)(fhi + j) * kT + t;
    float x0 = x[a];
    B0[j] = (_Float16)x0;                                  // Tx0 = x
    B1[j] = (_Float16)tx1[a];                              // Tx1 = L x
    B2[j] = (_Float16)(2.0f * tx2raw[a] - x0);             // Tx2 = 2 L Tx1 - Tx0
  }

  float8 acc[4] = {};
  const half16* ap = (const half16*)afrag;
#pragma unroll
  for (int m = 0; m < 4; ++m) {
    half16 a0 = ap[(0 * 4 + m) * 32 + lane];
    half16 a1 = ap[(1 * 4 + m) * 32 + lane];
    half16 a2 = ap[(2 * 4 + m) * 32 + lane];
    acc[m] = wmma_f16(a0, B0, acc[m]);
    acc[m] = wmma_f16(a1, B1, acc[m]);
    acc[m] = wmma_f16(a2, B2, acc[m]);
  }

  _Float16* sp = spatial + ((size_t)b * kN + n) * kNodeC;
  int mhi = (lane >= 16) ? 8 : 0;         // C/D layout: vgpr r -> M = r (+8 hi half)
#pragma unroll
  for (int m = 0; m < 4; ++m)
#pragma unroll
    for (int r = 0; r < 8; ++r) {
      int c = m * 16 + r + mhi;
      float v = acc[m][r] + chebB[c];
      v = v > 0.0f ? v : 0.0f;            // ReLU fused
      sp[c * kT + t] = (_Float16)v;       // f16 intermediate: halves HBM traffic
    }
}

// ------- temporal conv (K=3 over T) + 1x1 residual conv + ReLU + LayerNorm -------
__global__ void __launch_bounds__(128) time_res_ln_kernel(
    const float* __restrict__ x, const _Float16* __restrict__ spatial,
    const _Float16* __restrict__ tfrag, const _Float16* __restrict__ rfrag,
    const float* __restrict__ tb, const float* __restrict__ rb,
    const float* __restrict__ lg, const float* __restrict__ lbeta,
    float* __restrict__ out) {
  __shared__ _Float16 sp_s[4][kNodeC];    // 2 KB/wave
  __shared__ _Float16 xs_s[4][kNodeF];    // 1 KB/wave
  int wid = threadIdx.x >> 5, lane = threadIdx.x & 31;
  int node = blockIdx.x * 4 + wid;
  int b = node / kN, n = node % kN;

  // stage this node's activations in LDS (coalesced dword copies)
  const unsigned* spw = (const unsigned*)(spatial + ((size_t)b * kN + n) * kNodeC);
  unsigned* sdst = (unsigned*)sp_s[wid];
  for (int i = lane; i < kNodeC / 2; i += 32) sdst[i] = spw[i];
  const float* xb = x + ((size_t)b * kN + n) * kNodeF;
  for (int i = lane; i < kNodeF; i += 32) xs_s[wid][i] = (_Float16)xb[i];
  __syncthreads();

  int t = lane & 15;
  int khi = (lane >= 16) ? 16 : 0;

  float8 acc[4] = {};

  // residual 1x1 conv: (64x32) @ x(32x16)
  half16 Bx;
#pragma unroll
  for (int j = 0; j < 16; ++j) Bx[j] = xs_s[wid][(khi + j) * kT + t];
  const half16* rp = (const half16*)rfrag;
#pragma unroll
  for (int m = 0; m < 4; ++m) acc[m] = wmma_f16(rp[m * 32 + lane], Bx, acc[m]);

  // temporal conv: 3 shifted (64x64) @ sp(64x16), zero-padded at T edges
  const half16* tp = (const half16*)tfrag;
#pragma unroll
  for (int dt = 0; dt < 3; ++dt) {
    int ts = t + dt - 1;
    bool ok = (ts >= 0) && (ts < kT);     // per-lane select, no divergence
#pragma unroll
    for (int kt = 0; kt < 2; ++kt) {
      half16 Bs;
#pragma unroll
      for (int j = 0; j < 16; ++j) {
        int ci = kt * 32 + khi + j;
        Bs[j] = ok ? sp_s[wid][ci * kT + ts] : (_Float16)0.0f;
      }
#pragma unroll
      for (int m = 0; m < 4; ++m)
        acc[m] = wmma_f16(tp[((dt * 2 + kt) * 4 + m) * 32 + lane], Bs, acc[m]);
    }
  }

  // bias + ReLU + LayerNorm over the 64 channels of each (b,t,n)
  int mhi = (lane >= 16) ? 8 : 0;
  float z[4][8];
  float s1 = 0.0f, s2 = 0.0f;
#pragma unroll
  for (int m = 0; m < 4; ++m)
#pragma unroll
    for (int r = 0; r < 8; ++r) {
      int c = m * 16 + r + mhi;
      float v = acc[m][r] + tb[c] + rb[c];
      v = v > 0.0f ? v : 0.0f;
      z[m][r] = v;
      s1 += v;
      s2 += v * v;
    }
  // lanes L and L+16 together hold all 64 channels of column t -> one xor-16 hop
  s1 += __shfl_xor(s1, 16, 32);
  s2 += __shfl_xor(s2, 16, 32);
  float mu = s1 * (1.0f / 64.0f);
  float var = s2 * (1.0f / 64.0f) - mu * mu;
  float inv = rsqrtf(var + kEps);

  float* ob = out + ((size_t)b * kN + n) * kNodeC;
#pragma unroll
  for (int m = 0; m < 4; ++m)
#pragma unroll
    for (int r = 0; r < 8; ++r) {
      int c = m * 16 + r + mhi;
      ob[c * kT + t] = (z[m][r] - mu) * inv * lg[c] + lbeta[c];
    }
}

// ---------------- host ----------------
extern "C" void kernel_launch(void* const* d_in, const int* in_sizes, int n_in,
                              void* d_out, int out_size, void* d_ws, size_t ws_size,
                              hipStream_t stream) {
  (void)in_sizes; (void)n_in; (void)out_size; (void)ws_size;
  const float* x     = (const float*)d_in[0];
  const int*   ei    = (const int*)d_in[1];
  const float* chebW = (const float*)d_in[2];
  const float* chebB = (const float*)d_in[3];
  const float* timeW = (const float*)d_in[4];
  const float* timeB = (const float*)d_in[5];
  const float* resW  = (const float*)d_in[6];
  const float* resB  = (const float*)d_in[7];
  const float* lnG   = (const float*)d_in[8];
  const float* lnB   = (const float*)d_in[9];
  float* out = (float*)d_out;

  char* ws = (char*)d_ws;
  size_t off = 0;
  auto take = [&](size_t bytes) {
    char* p = ws + off;
    off = (off + bytes + 255) & ~(size_t)255;
    return p;
  };
  float*     deg     = (float*)take((size_t)kN * 4);
  float*     dinv    = (float*)take((size_t)kN * 4);
  float*     w       = (float*)take((size_t)kE * 4);
  float*     tx1     = (float*)take((size_t)kB * kN * kNodeF * 4);
  float*     tx2     = (float*)take((size_t)kB * kN * kNodeF * 4);
  _Float16*  spatial = (_Float16*)take((size_t)kB * kN * kNodeC * 2);
  _Float16*  chebF   = (_Float16*)take((size_t)3 * 4 * 512 * 2);
  _Float16*  timeF   = (_Float16*)take((size_t)3 * 2 * 4 * 512 * 2);
  _Float16*  resF    = (_Float16*)take((size_t)4 * 512 * 2);

  const long nTx = (long)kB * kN * kNodeF;

  zero_kernel<<<256, 256, 0, stream>>>(deg, kN);
  zero_kernel<<<4096, 256, 0, stream>>>(tx1, nTx);
  zero_kernel<<<4096, 256, 0, stream>>>(tx2, nTx);

  deg_kernel<<<(kE + 255) / 256, 256, 0, stream>>>(ei, deg);
  dinv_kernel<<<(kN + 255) / 256, 256, 0, stream>>>(deg, dinv);
  edgew_kernel<<<(kE + 255) / 256, 256, 0, stream>>>(ei, dinv, w);

  pack_cheb<<<(6144 + 255) / 256, 256, 0, stream>>>(chebW, chebF);
  pack_time<<<(12288 + 255) / 256, 256, 0, stream>>>(timeW, timeF);
  pack_res<<<(2048 + 255) / 256, 256, 0, stream>>>(resW, resF);

  prop_kernel<<<kE, 256, 0, stream>>>(x, tx1, w, ei, ei + kE);    // Tx1 = L x
  prop_kernel<<<kE, 256, 0, stream>>>(tx1, tx2, w, ei, ei + kE);  // raw L Tx1

  cheb_wmma_kernel<<<(kB * kN) / 4, 128, 0, stream>>>(x, tx1, tx2, chebF, chebB,
                                                      spatial);
  time_res_ln_kernel<<<(kB * kN) / 4, 128, 0, stream>>>(x, spatial, timeF, resF,
                                                        timeB, resB, lnG, lnB, out);
}
